// GRU_50044958933140
// MI455X (gfx1250) — matrix operand order repaired and checked by reading, not compiled, not run on hardware
//
#include <hip/hip_runtime.h>

// ---------------------------------------------------------------------------
// 2-layer GRU LM on gfx1250: bf16 WMMA (16x16x32) GEMMs, f32 accumulation.
//   T=128, B=64, V=10000, E=H=512, K=E+H=1024
// ---------------------------------------------------------------------------

typedef __bf16 v16bf __attribute__((ext_vector_type(16)));
typedef __bf16 v8bf  __attribute__((ext_vector_type(8)));
typedef float  v8f   __attribute__((ext_vector_type(8)));

#define T_  128
#define B_  64
#define V_  10000
#define E_  512
#define H_  512
#define K_  1024
#define LDA 1032   // padded LDS row stride in bf16 elems (2064B: 16B-aligned, bank-conflict free)

// A-fragment loader: lane holds row (lane&15); caller pre-offsets by (lane>>4)*16.
// elements 0..15 = K[k..k+15]  (contiguous k order, matches B fragments)
static __device__ __forceinline__ v16bf ldA(const __bf16* p) {
  v8bf lo = *(const v8bf*)(p);
  v8bf hi = *(const v8bf*)(p + 8);
  return __builtin_shufflevector(lo, hi, 0,1,2,3,4,5,6,7,8,9,10,11,12,13,14,15);
}

static __device__ __forceinline__ v8f wmma_bf16(v16bf a, v16bf b, v8f c) {
  return __builtin_amdgcn_wmma_f32_16x16x32_bf16(false, a, false, b, (short)0, c, false, false);
}

static __device__ __forceinline__ float sigmoid_f(float x) {
  return 1.0f / (1.0f + __expf(-x));
}
static __device__ __forceinline__ float tanh_fast(float x) {
  return 1.0f - 2.0f / (__expf(2.0f * x) + 1.0f);
}

// ---------------------------------------------------------------------------
// Embedding gather + f32 -> bf16 convert.  Exactly T*B*E = 4,194,304 threads.
// ---------------------------------------------------------------------------
__global__ __launch_bounds__(256) void embed_gather(
    const int* __restrict__ inputs, const float* __restrict__ emb,
    __bf16* __restrict__ x) {
  int idx = blockIdx.x * 256 + threadIdx.x;   // over T*B*E
  int tb  = idx >> 9;                          // / E_
  int e   = idx & (E_ - 1);
  x[idx] = (__bf16)emb[(size_t)inputs[tb] * E_ + e];
}

// ---------------------------------------------------------------------------
// Transpose-convert: in [K][N] f32  ->  out [N][K] bf16  (WMMA B-operand layout)
// ---------------------------------------------------------------------------
__global__ __launch_bounds__(256) void transpose_to_bf16(
    const float* __restrict__ in, __bf16* __restrict__ out, int K, int N) {
  int idx = blockIdx.x * 256 + threadIdx.x;
  if (idx < K * N) {
    int k = idx / N;
    int n = idx - k * N;
    out[(size_t)n * K + k] = (__bf16)in[idx];
  }
}

// ---------------------------------------------------------------------------
// Persistent GRU recurrence: one 32-wave workgroup runs all T steps.
//   LDS holds cat = [input(512) | h_or_r*h(512)] as bf16, row stride LDA.
//   Phase A: rz = sigmoid(cat @ Wrz^T + b)      (N = 1024, Wr|Wz stacked)
//   Phase B: cat[:,512:] = bf16(r * h_prev)
//   Phase C: h~ = tanh(cat @ Wh^T + bh);  h = (1-z)h + z h~
// Weight matrices are pre-transposed [N][K] bf16 so B-fragments are contiguous.
// K-loops: batched A-frag ds_loads + double-buffered B-frag global loads so the
// WMMA chain issues back-to-back after a single wait.
// ---------------------------------------------------------------------------
__global__ __launch_bounds__(1024) void gru_persist(
    const __bf16* __restrict__ xall,                          // [T][B][E]
    const __bf16* __restrict__ Wrz0t, const __bf16* __restrict__ Wh0t,
    const __bf16* __restrict__ Wrz1t, const __bf16* __restrict__ Wh1t,
    const float* __restrict__ br0, const float* __restrict__ bz0,
    const float* __restrict__ bh0,
    const float* __restrict__ br1, const float* __restrict__ bz1,
    const float* __restrict__ bh1,
    float* __restrict__ hst,                                  // [2][B][H] state
    float* __restrict__ rzbuf,                                // [B][1024] f32
    __bf16* __restrict__ h1all)                               // [T][B][H]
{
  extern __shared__ char smem[];
  __bf16* cat = (__bf16*)smem;                                // [64][LDA]

  const int tid  = threadIdx.x;
  const int wave = tid >> 5;
  const int lane = tid & 31;
  const int l16  = lane & 15;
  const int half = lane >> 4;

  float* h0 = hst;
  float* h1 = hst + B_ * H_;

  for (int t = 0; t < T_; ++t) {
    for (int layer = 0; layer < 2; ++layer) {
      const __bf16* Wrzt = layer ? Wrz1t : Wrz0t;
      const __bf16* Wht  = layer ? Wh1t  : Wh0t;
      const float*  br   = layer ? br1 : br0;
      const float*  bz   = layer ? bz1 : bz0;
      const float*  bh   = layer ? bh1 : bh0;
      float*        hprev = layer ? h1 : h0;
      const __bf16* xrow  = xall + (size_t)t * (B_ * E_);

      // ---- P1: build cat = [input | h_prev] in LDS --------------------
      if (layer == 0) {
        // Left half (x_t, already bf16): async global->LDS copy, no VGPR
        // round trip.  64 rows x 1024B = 4096 x 16B chunks.
        for (int c = tid; c < 4096; c += 1024) {
          int m  = c >> 6;
          int kb = (c & 63) << 4;                 // byte offset within row
          unsigned lds_addr =
              (unsigned)(size_t)((char*)cat + m * (LDA * 2) + kb);
          const char* gp = (const char*)xrow + m * 1024 + kb;
          asm volatile("global_load_async_to_lds_b128 %0, %1, off"
                       :: "v"(lds_addr), "v"(gp) : "memory");
        }
        // Right half: f32 h0 -> bf16
        for (int idx = tid; idx < B_ * H_; idx += 1024) {
          int m = idx >> 9, k = idx & (H_ - 1);
          cat[m * LDA + 512 + k] = (__bf16)h0[m * H_ + k];
        }
        asm volatile("s_wait_asynccnt 0" ::: "memory");
      } else {
        for (int idx = tid; idx < B_ * K_; idx += 1024) {
          int m = idx >> 10;
          int k = idx & (K_ - 1);
          __bf16 v;
          if (k < 512) v = (__bf16)h0[m * H_ + k];
          else         v = (__bf16)h1[m * H_ + (k - 512)];
          cat[m * LDA + k] = v;
        }
      }
      __syncthreads();

      // ---- P2: rz GEMM, N=1024 (2 col-blocks per wave, 4 row-blocks) --
      {
        const __bf16* a0 = cat + l16 * LDA + half * 16;
        for (int cbi = 0; cbi < 2; ++cbi) {
          const int n = (wave * 2 + cbi) * 16 + l16;
          const __bf16* wrow = Wrzt + (size_t)n * K_ + half * 16;
          v8f acc[4] = {};
          v16bf bnext = *(const v16bf*)(wrow);
          for (int ks = 0; ks < 32; ++ks) {
            v16bf bcur = bnext;
            bnext = *(const v16bf*)(wrow + (ks + 1) * 32);  // <=64B overrun stays in ws
            __builtin_prefetch(wrow + (ks + 2) * 32, 0, 1);
            const __bf16* ap = a0 + ks * 32;
            v16bf afr[4];
            #pragma unroll
            for (int mb = 0; mb < 4; ++mb)
              afr[mb] = ldA(ap + mb * 16 * LDA);
            #pragma unroll
            for (int mb = 0; mb < 4; ++mb)
              acc[mb] = wmma_bf16(afr[mb], bcur, acc[mb]);
          }
          const float bias = (n < 512) ? br[n] : bz[n - 512];
          #pragma unroll
          for (int mb = 0; mb < 4; ++mb) {
            #pragma unroll
            for (int j = 0; j < 8; ++j) {
              int row = mb * 16 + half * 8 + j;
              rzbuf[row * 1024 + n] = sigmoid_f(acc[mb][j] + bias);
            }
          }
        }
      }
      __threadfence();
      __syncthreads();

      // ---- P3: cat right half <- bf16(r * h_prev) ---------------------
      for (int idx = tid; idx < B_ * H_; idx += 1024) {
        int m  = idx >> 9;
        int nn = idx & (H_ - 1);
        cat[m * LDA + 512 + nn] =
            (__bf16)(rzbuf[m * 1024 + nn] * hprev[m * H_ + nn]);
      }
      __syncthreads();

      // ---- P4: h~ GEMM, N=512 (1 col-block per wave) + gate fuse ------
      {
        const __bf16* a0 = cat + l16 * LDA + half * 16;
        const int n = wave * 16 + l16;
        const __bf16* wrow = Wht + (size_t)n * K_ + half * 16;
        v8f acc[4] = {};
        v16bf bnext = *(const v16bf*)(wrow);
        for (int ks = 0; ks < 32; ++ks) {
          v16bf bcur = bnext;
          bnext = *(const v16bf*)(wrow + (ks + 1) * 32);
          __builtin_prefetch(wrow + (ks + 2) * 32, 0, 1);
          const __bf16* ap = a0 + ks * 32;
          v16bf afr[4];
          #pragma unroll
          for (int mb = 0; mb < 4; ++mb)
            afr[mb] = ldA(ap + mb * 16 * LDA);
          #pragma unroll
          for (int mb = 0; mb < 4; ++mb)
            acc[mb] = wmma_bf16(afr[mb], bcur, acc[mb]);
        }
        const float bias = bh[n];
        #pragma unroll
        for (int mb = 0; mb < 4; ++mb) {
          #pragma unroll
          for (int j = 0; j < 8; ++j) {
            int row  = mb * 16 + half * 8 + j;
            float ht = tanh_fast(acc[mb][j] + bias);
            float z  = rzbuf[row * 1024 + 512 + n];
            float hp = hprev[row * H_ + n];
            float hn = (1.0f - z) * hp + z * ht;
            hprev[row * H_ + n] = hn;
            if (layer) h1all[((size_t)t * B_ + row) * H_ + n] = (__bf16)hn;
          }
        }
      }
      __threadfence();
      __syncthreads();
    }
  }
}

// ---------------------------------------------------------------------------
// Output projection: [8192][512]bf16 @ [512][10000] + by -> f32 logits.
// Block = 8 waves = 128 rows x 80 cols.  Grid (125, 64).
// ---------------------------------------------------------------------------
__global__ __launch_bounds__(256) void logits_gemm(
    const __bf16* __restrict__ A,     // [T*B][H] bf16
    const __bf16* __restrict__ Wyt,   // [V][H]  bf16 (transposed)
    const float* __restrict__ by,
    float* __restrict__ out)          // [T*B][V] f32
{
  const int wave = threadIdx.x >> 5;
  const int lane = threadIdx.x & 31;
  const int l16  = lane & 15;
  const int half = lane >> 4;
  const int m0   = blockIdx.y * 128 + wave * 16;
  const int n0   = blockIdx.x * 80;

  const __bf16* ap = A + (size_t)(m0 + l16) * H_ + half * 16;
  v8f acc[5] = {};
  for (int ks = 0; ks < 16; ++ks) {
    v16bf af = ldA(ap + ks * 32);
    v16bf bfr[5];
    #pragma unroll
    for (int j = 0; j < 5; ++j) {
      int n = n0 + j * 16 + l16;
      bfr[j] = *(const v16bf*)(Wyt + (size_t)n * H_ + ks * 32 + half * 16);
    }
    #pragma unroll
    for (int j = 0; j < 5; ++j)
      acc[j] = wmma_bf16(af, bfr[j], acc[j]);
  }
  #pragma unroll
  for (int j = 0; j < 5; ++j) {
    int n = n0 + j * 16 + l16;
    float bias = by[n];
    #pragma unroll
    for (int r = 0; r < 8; ++r) {
      int row = m0 + half * 8 + r;
      out[(size_t)row * V_ + n] = acc[j][r] + bias;
    }
  }
}

// ---------------------------------------------------------------------------
extern "C" void kernel_launch(void* const* d_in, const int* in_sizes, int n_in,
                              void* d_out, int out_size, void* d_ws, size_t ws_size,
                              hipStream_t stream) {
  (void)in_sizes; (void)n_in; (void)out_size; (void)ws_size;

  const int*   inputs = (const int*)  d_in[0];
  const float* hidden = (const float*)d_in[1];
  const float* emb    = (const float*)d_in[2];
  const float* Wr0 = (const float*)d_in[3];  const float* br0 = (const float*)d_in[4];
  const float* Wz0 = (const float*)d_in[5];  const float* bz0 = (const float*)d_in[6];
  const float* Wh0 = (const float*)d_in[7];  const float* bh0 = (const float*)d_in[8];
  const float* Wr1 = (const float*)d_in[9];  const float* br1 = (const float*)d_in[10];
  const float* Wz1 = (const float*)d_in[11]; const float* bz1 = (const float*)d_in[12];
  const float* Wh1 = (const float*)d_in[13]; const float* bh1 = (const float*)d_in[14];
  const float* Wy  = (const float*)d_in[15]; const float* by  = (const float*)d_in[16];
  float* out = (float*)d_out;

  char* ws = (char*)d_ws;
  __bf16* xall  = (__bf16*)(ws);               //  8,388,608 B  [T][B][E] bf16
  __bf16* Wrz0t = (__bf16*)(ws + 8388608);     //  2,097,152 B  [1024][1024] (Wr0^T|Wz0^T)
  __bf16* Wh0t  = (__bf16*)(ws + 10485760);    //  1,048,576 B  [512][1024]
  __bf16* Wrz1t = (__bf16*)(ws + 11534336);    //  2,097,152 B
  __bf16* Wh1t  = (__bf16*)(ws + 13631488);    //  1,048,576 B
  __bf16* Wyt   = (__bf16*)(ws + 14680064);    // 10,240,000 B  [10000][512]
  float*  hst   = (float*)(ws + 24920064);     //    524,288 B  [2][64][512]
  float*  rzbuf = (float*)(ws + 25444352);     //    262,144 B  [64][1024]
  __bf16* h1all = (__bf16*)(ws + 25706496);    //  8,388,608 B  [T][B][H]
                                               // total ~34.1 MB (L2-resident)

  // 1) embedding -> bf16
  embed_gather<<<16384, 256, 0, stream>>>(inputs, emb, xall);

  // 2) weights -> bf16, transposed to [N][K] (B-operand layout); Wr|Wz stacked
  transpose_to_bf16<<<2048, 256, 0, stream>>>(Wr0, Wrz0t,              1024, 512);
  transpose_to_bf16<<<2048, 256, 0, stream>>>(Wz0, Wrz0t + 512 * 1024, 1024, 512);
  transpose_to_bf16<<<2048, 256, 0, stream>>>(Wh0, Wh0t,               1024, 512);
  transpose_to_bf16<<<2048, 256, 0, stream>>>(Wr1, Wrz1t,              1024, 512);
  transpose_to_bf16<<<2048, 256, 0, stream>>>(Wz1, Wrz1t + 512 * 1024, 1024, 512);
  transpose_to_bf16<<<2048, 256, 0, stream>>>(Wh1, Wh1t,               1024, 512);
  transpose_to_bf16<<<20000, 256, 0, stream>>>(Wy, Wyt, 512, 10000);

  // 3) initial hidden state
  hipMemcpyAsync(hst, hidden, 2 * B_ * H_ * sizeof(float),
                 hipMemcpyDeviceToDevice, stream);

  // 4) persistent recurrence (one 32-wave workgroup, 129KB dynamic LDS)
  gru_persist<<<1, 1024, 64 * LDA * (int)sizeof(__bf16), stream>>>(
      xall, Wrz0t, Wh0t, Wrz1t, Wh1t,
      br0, bz0, bh0, br1, bz1, bh1,
      hst, rzbuf, h1all);

  // 5) vocab projection over all timesteps at once
  logits_gemm<<<dim3(125, 64), 256, 0, stream>>>(h1all, Wyt, by, out);

  // 6) final hidden -> tail of d_out
  hipMemcpyAsync(out + (size_t)T_ * B_ * V_, hst,
                 2 * B_ * H_ * sizeof(float), hipMemcpyDeviceToDevice, stream);
}
